// DisGraphRep_14448269984333
// MI455X (gfx1250) — compile-verified, matrix operand order checked by reading, not run on hardware
//
#include <hip/hip_runtime.h>
#include <hip/hip_bf16.h>

#define HID   128
#define NPOI  50000
#define NEDGE 800000
#define NVAL  ((size_t)NPOI * HID)

typedef __attribute__((ext_vector_type(16))) _Float16 v16h;
typedef __attribute__((ext_vector_type(8)))  _Float16 v8h;
typedef __attribute__((ext_vector_type(8)))  float    v8f;
typedef __attribute__((ext_vector_type(4)))  float    v4f;

// ---------------------------------------------------------------------------
// fp32 -> f16 transposed weight convert: Wt[n*128+k] = (f16) W[k*128+n]
// ---------------------------------------------------------------------------
__global__ __launch_bounds__(256) void convert_wt_kernel(
    const float* __restrict__ W, _Float16* __restrict__ Wt) {
  int t = blockIdx.x * 256 + threadIdx.x;      // 16384 total
  int n = t >> 7, k = t & 127;
  Wt[t] = (_Float16)W[k * HID + n];
}

// ---------------------------------------------------------------------------
// WMMA GEMM: Out[M x 128] = A[M x 128] @ W (+bias) (+resid) (relu?)
// Wt is f16, transposed: Wt[n][k]. Block = 256 thr = 8 waves, tile 128 rows.
// Weights staged once per block into LDS (32 KB), b-frags via ds_load_b128.
// ---------------------------------------------------------------------------
__global__ __launch_bounds__(256) void gemm128_wmma_kernel(
    const float* __restrict__ A, const _Float16* __restrict__ Wt,
    const float* __restrict__ bias, const float* __restrict__ resid,
    float* __restrict__ Out, int M, int doRelu) {
  __shared__ _Float16 wlds[HID * HID];         // 32 KB

  // cooperative stage: 16384 halves = 2048 x 16B chunks, 8 per thread
  {
    const v8h* wsrc = (const v8h*)Wt;
    v8h* wdst = (v8h*)wlds;
#pragma unroll
    for (int i = 0; i < 8; ++i)
      wdst[threadIdx.x + i * 256] = wsrc[threadIdx.x + i * 256];
  }
  __syncthreads();

  const int lane = threadIdx.x & 31;
  const int wave = threadIdx.x >> 5;
  const int rl   = lane & 15;           // row/col within 16-tile
  const int kh   = (lane >> 4) * 8;     // K sub-chunk select (wave32 layout)
  const int rowBase = blockIdx.x * 128 + wave * 16;
  int rclamp = rowBase + rl;
  if (rclamp >= M) rclamp = M - 1;      // clamp loads; stores are guarded

  v8f acc[8] = {};

#pragma unroll
  for (int kt = 0; kt < 4; ++kt) {
    const int kb = kt * 32;
    // A fragment 16x32 f16: lane<16 -> K {kb..kb+7, kb+16..kb+23}
    const float* ap = A + (size_t)rclamp * HID + kb + kh;
    v16h afrag;
#pragma unroll
    for (int j = 0; j < 8; ++j) afrag[j] = (_Float16)ap[j];
#pragma unroll
    for (int j = 0; j < 8; ++j) afrag[8 + j] = (_Float16)ap[16 + j];

#pragma unroll
    for (int nt = 0; nt < 8; ++nt) {
      const _Float16* bp = wlds + (nt * 16 + rl) * HID + kb + kh;
      v8h b0 = *(const v8h*)(bp);
      v8h b1 = *(const v8h*)(bp + 16);
      v16h bfrag;
#pragma unroll
      for (int j = 0; j < 8; ++j) { bfrag[j] = b0[j]; bfrag[8 + j] = b1[j]; }
      acc[nt] = __builtin_amdgcn_wmma_f32_16x16x32_f16(
          false, afrag, false, bfrag, (short)0, acc[nt], false, false);
    }
  }

  // C/D layout: VGPR v -> row = base + (lane>=16?8:0) + v ; col = nt*16 + (lane&15)
  const int rOff = (lane >> 4) * 8;
#pragma unroll
  for (int nt = 0; nt < 8; ++nt) {
    const int c = nt * 16 + rl;
#pragma unroll
    for (int v = 0; v < 8; ++v) {
      int r = rowBase + rOff + v;
      if (r < M) {
        float val = acc[nt][v];
        if (bias)  val += bias[c];
        if (resid) val += resid[(size_t)r * HID + c];
        if (doRelu) val = val > 0.0f ? val : 0.0f;
        Out[(size_t)r * HID + c] = val;
      }
    }
  }
}

// ---------------------------------------------------------------------------
// Row LayerNorm over 128 features: one wave per row (wave32: 4 elems/lane)
// ---------------------------------------------------------------------------
__global__ __launch_bounds__(256) void ln_kernel(
    const float* __restrict__ X, const float* __restrict__ g,
    const float* __restrict__ b, float* __restrict__ Out, int M) {
  int lane = threadIdx.x & 31;
  int row  = blockIdx.x * 8 + (threadIdx.x >> 5);
  if (row >= M) return;
  v4f x = *(const v4f*)(X + (size_t)row * HID + lane * 4);
  float s  = x[0] + x[1] + x[2] + x[3];
  float sq = x[0]*x[0] + x[1]*x[1] + x[2]*x[2] + x[3]*x[3];
  for (int m = 16; m >= 1; m >>= 1) { s += __shfl_xor(s, m, 32); sq += __shfl_xor(sq, m, 32); }
  float mean = s * (1.0f / 128.0f);
  float var  = sq * (1.0f / 128.0f) - mean * mean;
  float rstd = rsqrtf(var + 1e-5f);
  float* op = Out + (size_t)row * HID + lane * 4;
#pragma unroll
  for (int j = 0; j < 4; ++j)
    op[j] = (x[j] - mean) * rstd * g[lane * 4 + j] + b[lane * 4 + j];
}

// out2 = LN(a+b); xnext = leaky(out2); outsum += xnext
__global__ __launch_bounds__(256) void add_ln_leaky_kernel(
    const float* __restrict__ A, const float* __restrict__ B,
    const float* __restrict__ g, const float* __restrict__ bb,
    float* __restrict__ Xnext, float* __restrict__ Outsum, int M) {
  int lane = threadIdx.x & 31;
  int row  = blockIdx.x * 8 + (threadIdx.x >> 5);
  if (row >= M) return;
  v4f a = *(const v4f*)(A + (size_t)row * HID + lane * 4);
  v4f c = *(const v4f*)(B + (size_t)row * HID + lane * 4);
  v4f x;
#pragma unroll
  for (int j = 0; j < 4; ++j) x[j] = a[j] + c[j];
  float s  = x[0] + x[1] + x[2] + x[3];
  float sq = x[0]*x[0] + x[1]*x[1] + x[2]*x[2] + x[3]*x[3];
  for (int m = 16; m >= 1; m >>= 1) { s += __shfl_xor(s, m, 32); sq += __shfl_xor(sq, m, 32); }
  float mean = s * (1.0f / 128.0f);
  float var  = sq * (1.0f / 128.0f) - mean * mean;
  float rstd = rsqrtf(var + 1e-5f);
  float* xp = Xnext  + (size_t)row * HID + lane * 4;
  float* op = Outsum + (size_t)row * HID + lane * 4;
#pragma unroll
  for (int j = 0; j < 4; ++j) {
    float y = (x[j] - mean) * rstd * g[lane * 4 + j] + bb[lane * 4 + j];
    y = y > 0.0f ? y : 0.01f * y;       // LeakyReLU
    xp[j] = y;
    op[j] += y;
  }
}

// ---------------------------------------------------------------------------
// init: agg = 0, segmax = INT_MIN, denom = 0
// ---------------------------------------------------------------------------
__global__ __launch_bounds__(256) void init_edge_state_kernel(
    float* __restrict__ agg, int* __restrict__ segmax, float* __restrict__ denom) {
  size_t t = (size_t)blockIdx.x * 256 + threadIdx.x;
  if (t < NVAL) agg[t] = 0.0f;
  if (t < 2 * NPOI) { segmax[t] = (int)0x80000000; denom[t] = 0.0f; }
}

__device__ __forceinline__ int fmap_mono(float f) {
  int i = __float_as_int(f);
  return i >= 0 ? i : (i ^ 0x7fffffff);
}
__device__ __forceinline__ float fmap_inv(int i) {
  return __int_as_float(i >= 0 ? i : (i ^ 0x7fffffff));
}

// Pass 1: logits[e,h] = dot(Q[dst],K[src]) per head /8, atomic seg-max. Wave/edge.
__global__ __launch_bounds__(256) void edge_logits_kernel(
    const int* __restrict__ src, const int* __restrict__ dst,
    const float* __restrict__ Q, const float* __restrict__ K,
    float* __restrict__ logits, int* __restrict__ segmax, int E) {
  int lane = threadIdx.x & 31;
  int e = blockIdx.x * 8 + (threadIdx.x >> 5);
  if (e >= E) return;
  if (lane == 0) {                      // stream-prefetch the sequential index arrays
    __builtin_prefetch(src + e + 256, 0, 1);
    __builtin_prefetch(dst + e + 256, 0, 1);
  }
  int d = dst[e], s = src[e];
  v4f q = *(const v4f*)(Q + (size_t)d * HID + lane * 4);
  v4f k = *(const v4f*)(K + (size_t)s * HID + lane * 4);
  float p = q[0]*k[0] + q[1]*k[1] + q[2]*k[2] + q[3]*k[3];
  for (int m = 8; m >= 1; m >>= 1) p += __shfl_xor(p, m, 32);  // per-16-lane head
  float logit = p * 0.125f;                                    // 1/sqrt(64)
  if ((lane & 15) == 0) {
    int h = lane >> 4;
    logits[(size_t)e * 2 + h] = logit;
    atomicMax(&segmax[d * 2 + h], fmap_mono(logit));
  }
}

// Pass 2: ex = exp(logit - segmax[dst]); denom[dst] += ex. Thread per (edge,head).
__global__ __launch_bounds__(256) void edge_expsum_kernel(
    const int* __restrict__ dst, float* __restrict__ logits,
    const int* __restrict__ segmax, float* __restrict__ denom, int E) {
  int t = blockIdx.x * 256 + threadIdx.x;
  if (t >= 2 * E) return;
  int e = t >> 1, h = t & 1;
  int d = dst[e];
  float smax = fmap_inv(segmax[d * 2 + h]);
  float ex = __expf(logits[t] - smax);
  logits[t] = ex;                                 // in-place: logits -> ex
  __hip_atomic_fetch_add(&denom[d * 2 + h], ex,
                         __ATOMIC_RELAXED, __HIP_MEMORY_SCOPE_AGENT);
}

// Pass 3: agg[dst] += (ex/denom) * V[src]. Wave per edge, f32 L2 atomics.
__global__ __launch_bounds__(256) void edge_agg_kernel(
    const int* __restrict__ src, const int* __restrict__ dst,
    const float* __restrict__ ex, const float* __restrict__ denom,
    const float* __restrict__ V, float* __restrict__ agg, int E) {
  int lane = threadIdx.x & 31;
  int e = blockIdx.x * 8 + (threadIdx.x >> 5);
  if (e >= E) return;
  if (lane == 0) {
    __builtin_prefetch(src + e + 256, 0, 1);
    __builtin_prefetch(ex + 2 * (e + 256), 0, 1);
  }
  int d = dst[e], s = src[e];
  int h = lane >> 4;                               // elems 0-63 head0, 64-127 head1
  float alpha = ex[(size_t)e * 2 + h] / (denom[d * 2 + h] + 1e-16f);
  v4f v = *(const v4f*)(V + (size_t)s * HID + lane * 4);
  float* ap = agg + (size_t)d * HID + lane * 4;
#pragma unroll
  for (int j = 0; j < 4; ++j)
    __hip_atomic_fetch_add(ap + j, alpha * v[j],
                           __ATOMIC_RELAXED, __HIP_MEMORY_SCOPE_AGENT);
}

__global__ __launch_bounds__(256) void scale_kernel(float* __restrict__ out, float sc) {
  size_t t = (size_t)blockIdx.x * 256 + threadIdx.x;
  if (t < NVAL) out[t] *= sc;
}

// ---------------------------------------------------------------------------
extern "C" void kernel_launch(void* const* d_in, const int* in_sizes, int n_in,
                              void* d_out, int out_size, void* d_ws, size_t ws_size,
                              hipStream_t stream) {
  (void)in_sizes; (void)n_in; (void)out_size; (void)ws_size;
  const float* x0  = (const float*)d_in[0];
  const int*  eidx = (const int*)d_in[1];
  const int*  src  = eidx;           // edge_index[0] = source j
  const int*  dst  = eidx + NEDGE;   // edge_index[1] = target i
  // param dict insertion order per layer (25 arrays), base index 3:
  // 0 WQ 1 WK 2 WV 3 WO 4 bO 5 WE 6 bE 7 WOe 8 bOe 9 Wf1 10 bf1 11 Wf2 12 bf2
  // 13 We1 14 be1 15 We2 16 be2 17 g1 18 b1 19 g2 20 b2 21 g1e 22 b1e 23 g2e 24 b2e
  auto P = [&](int l, int i) -> const float* { return (const float*)d_in[3 + l * 25 + i]; };

  // workspace carve (floats)
  float* ws = (float*)d_ws;
  size_t off = 0;
  float* bx   = ws + off; off += NVAL;       // current x (after layer 0)
  float* bq   = ws + off; off += NVAL;       // Q / out1
  float* bk   = ws + off; off += NVAL;       // K / out1ln
  float* bv   = ws + off; off += NVAL;       // V / h
  float* bagg = ws + off; off += NVAL;       // agg / h2
  float* elog = ws + off; off += 2 * (size_t)NEDGE;  // logits -> ex (in-place)
  int*   smax = (int*)(ws + off); off += 2 * NPOI;
  float* deno = ws + off; off += 2 * NPOI;
  _Float16* wt16 = (_Float16*)(ws + off);    // 12 * 16384 f16

  // convert + transpose all 6 GEMM weights per layer to f16
  const int widx[6] = {0, 1, 2, 3, 9, 11};   // WQ WK WV WO Wf1 Wf2
  for (int l = 0; l < 2; ++l)
    for (int j = 0; j < 6; ++j)
      convert_wt_kernel<<<64, 256, 0, stream>>>(P(l, widx[j]),
                                                wt16 + (size_t)(l * 6 + j) * 16384);

  // outsum starts as x0
  hipMemcpyAsync(d_out, (const void*)x0, NVAL * sizeof(float),
                 hipMemcpyDeviceToDevice, stream);

  const int GEMM_GRID = (NPOI + 127) / 128;       // 391
  const int EDGE_GRID = (NEDGE + 7) / 8;          // 100000
  const int ROW_GRID  = NPOI / 8;                 // 6250
  const int ELT_GRID  = (int)(NVAL / 256);        // 25000

  const float* xcur = x0;
  for (int l = 0; l < 2; ++l) {
    const _Float16* WQt = wt16 + (size_t)(l * 6 + 0) * 16384;
    const _Float16* WKt = wt16 + (size_t)(l * 6 + 1) * 16384;
    const _Float16* WVt = wt16 + (size_t)(l * 6 + 2) * 16384;
    const _Float16* WOt = wt16 + (size_t)(l * 6 + 3) * 16384;
    const _Float16* W1t = wt16 + (size_t)(l * 6 + 4) * 16384;
    const _Float16* W2t = wt16 + (size_t)(l * 6 + 5) * 16384;

    gemm128_wmma_kernel<<<GEMM_GRID, 256, 0, stream>>>(xcur, WQt, nullptr, nullptr, bq, NPOI, 0);
    gemm128_wmma_kernel<<<GEMM_GRID, 256, 0, stream>>>(xcur, WKt, nullptr, nullptr, bk, NPOI, 0);
    gemm128_wmma_kernel<<<GEMM_GRID, 256, 0, stream>>>(xcur, WVt, nullptr, nullptr, bv, NPOI, 0);

    init_edge_state_kernel<<<ELT_GRID, 256, 0, stream>>>(bagg, smax, deno);
    edge_logits_kernel<<<EDGE_GRID, 256, 0, stream>>>(src, dst, bq, bk, elog, smax, NEDGE);
    edge_expsum_kernel<<<(2 * NEDGE) / 256, 256, 0, stream>>>(dst, elog, smax, deno, NEDGE);
    edge_agg_kernel<<<EDGE_GRID, 256, 0, stream>>>(src, dst, elog, deno, bv, bagg, NEDGE);

    // out1 = agg@WO + bO + x ; out1ln = LN(out1)
    gemm128_wmma_kernel<<<GEMM_GRID, 256, 0, stream>>>(bagg, WOt, P(l, 4), xcur, bq, NPOI, 0);
    ln_kernel<<<ROW_GRID, 256, 0, stream>>>(bq, P(l, 17), P(l, 18), bk, NPOI);
    // h = relu(out1ln@Wf1 + bf1) ; h2 = h@Wf2 + bf2
    gemm128_wmma_kernel<<<GEMM_GRID, 256, 0, stream>>>(bk, W1t, P(l, 10), nullptr, bv, NPOI, 1);
    gemm128_wmma_kernel<<<GEMM_GRID, 256, 0, stream>>>(bv, W2t, P(l, 12), nullptr, bagg, NPOI, 0);
    // x = leaky(LN(out1ln + h2)); outsum += x
    add_ln_leaky_kernel<<<ROW_GRID, 256, 0, stream>>>(bk, bagg, P(l, 19), P(l, 20),
                                                      bx, (float*)d_out, NPOI);
    xcur = bx;
  }

  scale_kernel<<<ELT_GRID, 256, 0, stream>>>((float*)d_out, 1.0f / 3.0f);
}
